// Conv2dDeformable_14929306321759
// MI455X (gfx1250) — matrix operand order repaired and checked by reading, not compiled, not run on hardware
//
#include <hip/hip_runtime.h>
#include <hip/hip_bf16.h>

typedef _Float16 half4  __attribute__((ext_vector_type(4)));
typedef _Float16 half8  __attribute__((ext_vector_type(8)));
typedef _Float16 half16 __attribute__((ext_vector_type(16)));
typedef float    floatx8 __attribute__((ext_vector_type(8)));

#define KTOT 576   // 9 * 64 (k = (kh*3+kw)*64 + ci)

// ---------------------------------------------------------------------------
// Pack conv weights (OIHW f32) into f16 N-major [co][k] matrices so that a
// WMMA B fragment (16 K of one output-channel column, two runs of 8 at 8h and 16+8h) is two
// contiguous b128 loads per lane.
// ---------------------------------------------------------------------------
__global__ __launch_bounds__(256) void pack_weights(
    const float* __restrict__ w_off, const float* __restrict__ w_reg,
    _Float16* __restrict__ wofft,    // [128][576]
    _Float16* __restrict__ wregt)    // [ 64][576]
{
  int j = blockIdx.x * 256 + threadIdx.x;
  const int N1 = 128 * KTOT;
  if (j < N1) {
    int co = j / KTOT, k = j - co * KTOT;
    int khw = k >> 6, ci = k & 63;
    int kh = khw / 3, kw = khw - kh * 3;
    wofft[j] = (_Float16)w_off[((co * 64 + ci) * 3 + kh) * 3 + kw];
  } else if (j < N1 + 64 * KTOT) {
    int jj = j - N1;
    int co = jj / KTOT, k = jj - co * KTOT;
    int khw = k >> 6, ci = k & 63;
    int kh = khw / 3, kw = khw - kh * 3;
    wregt[jj] = (_Float16)w_reg[((co * 64 + ci) * 3 + kh) * 3 + kw];
  }
}

// ---------------------------------------------------------------------------
// x (NCHW f32) -> xh (NHWC f16): channel-contiguous im2col A tiles.
// One half4 (4 channels) per thread; w fastest for coalesced reads.
// ---------------------------------------------------------------------------
__global__ __launch_bounds__(256) void convert_x(
    const float* __restrict__ x, _Float16* __restrict__ xh)
{
  int j  = blockIdx.x * 256 + threadIdx.x;   // 2^21 chunks total
  int w  = j & 127;
  int cg = (j >> 7) & 15;
  int h  = (j >> 11) & 127;
  int b  = j >> 18;
  size_t in0 = ((size_t)(b * 64 + cg * 4) * 128 + h) * 128 + w;
  half4 v;
  v.x = (_Float16)x[in0];
  v.y = (_Float16)x[in0 + 16384];
  v.z = (_Float16)x[in0 + 32768];
  v.w = (_Float16)x[in0 + 49152];
  *(half4*)&xh[(((size_t)b * 128 + h) * 128 + w) * 64 + cg * 4] = v;
}

__device__ __forceinline__ float gather_px(const float* __restrict__ img,
                                           int xi, int yi)
{
  if (xi < 0 || xi > 127 || yi < 0 || yi > 127) return 0.0f;
  return img[yi * 128 + xi];
}

// A fragment: lane l holds row m = l&15; K runs [8*(l>>4), +8) and +16.
#define AFRAG(f)                                                              \
  ({                                                                          \
    const _Float16* a_ =                                                      \
        &atile[(kh * 66 + 16 * (f) + l15 + kw) * 64 + cib];                   \
    half8 alo_ = *(const half8*)a_;                                           \
    half8 ahi_ = *(const half8*)(a_ + 16);                                    \
    __builtin_shufflevector(alo_, ahi_, 0, 1, 2, 3, 4, 5, 6, 7, 8, 9, 10,     \
                            11, 12, 13, 14, 15);                              \
  })

// ---------------------------------------------------------------------------
// Fused: offset conv (implicit GEMM, WMMA f16->f32) + bilinear grid sample.
// WG = 8 waves; M-tile = 64 pixels (one row segment), wave n owns out
// channels [16n,16n+16) -> all 128 offset channels resident per WG.
// ---------------------------------------------------------------------------
__global__ __launch_bounds__(256) void conv_offset_sample(
    const float* __restrict__ x,        // f32 NCHW (for gathers)
    const _Float16* __restrict__ xh,    // f16 NHWC
    const _Float16* __restrict__ wofft, // [128][576]
    _Float16* __restrict__ xs)          // f16 NHWC sampled output
{
  __shared__ _Float16 atile[3 * 66 * 64] __attribute__((aligned(16)));
  __shared__ float off_lds[128 * 65];   // stride 65: bank-conflict-free

  const int tid  = threadIdx.x;
  const int wid  = tid >> 5;
  const int lane = tid & 31;
  const int l15  = lane & 15;
  const int lh   = lane >> 4;

  const int gid = blockIdx.x;           // b*256 + h*2 + wt
  const int wt  = gid & 1;
  const int h   = (gid >> 1) & 127;
  const int b   = gid >> 8;
  const int w0  = wt * 64;

  // ---- stage A window: rows h-1..h+1, px w0-1..w0+64, all 64 channels ----
  for (int t = tid; t < 3 * 66 * 16; t += 256) {
    int g = t & 15;
    int p = (t >> 4) % 66;
    int r = (t >> 4) / 66;
    int hh = h - 1 + r, ww = w0 - 1 + p;
    half4 v = {};
    if (hh >= 0 && hh < 128 && ww >= 0 && ww < 128)
      v = *(const half4*)&xh[(((size_t)b * 128 + hh) * 128 + ww) * 64 + g * 4];
    *(half4*)&atile[(r * 66 + p) * 64 + g * 4] = v;
  }
  __syncthreads();

  // ---- implicit GEMM: 64(M) x 16(N) per wave, K = 576 ----
  floatx8 acc0 = {}, acc1 = {}, acc2 = {}, acc3 = {};
  const int n0 = wid * 16;
  const _Float16* bptr = wofft + (size_t)(n0 + l15) * KTOT + 8 * lh;

  for (int s = 0; s < 18; ++s) {
    int khw = s >> 1;
    int kh = khw / 3, kw = khw - kh * 3;
    int cib = (s & 1) * 32 + 8 * lh;

    half8 blo = *(const half8*)(bptr + 32 * s);
    half8 bhi = *(const half8*)(bptr + 32 * s + 16);
    half16 Bf = __builtin_shufflevector(blo, bhi, 0, 1, 2, 3, 4, 5, 6, 7, 8,
                                        9, 10, 11, 12, 13, 14, 15);

    acc0 = __builtin_amdgcn_wmma_f32_16x16x32_f16(false, AFRAG(0), false, Bf,
                                                  (short)0, acc0, false, false);
    acc1 = __builtin_amdgcn_wmma_f32_16x16x32_f16(false, AFRAG(1), false, Bf,
                                                  (short)0, acc1, false, false);
    acc2 = __builtin_amdgcn_wmma_f32_16x16x32_f16(false, AFRAG(2), false, Bf,
                                                  (short)0, acc2, false, false);
    acc3 = __builtin_amdgcn_wmma_f32_16x16x32_f16(false, AFRAG(3), false, Bf,
                                                  (short)0, acc3, false, false);
  }

  // ---- offsets -> LDS (D layout: m = 16f + r + 8*(lane>>4), n = lane&15) ----
  {
    int n  = wid * 16 + l15;
    int m0 = 8 * lh;
#pragma unroll
    for (int r = 0; r < 8; ++r) {
      off_lds[n * 65 +  0 + m0 + r] = acc0[r];
      off_lds[n * 65 + 16 + m0 + r] = acc1[r];
      off_lds[n * 65 + 32 + m0 + r] = acc2[r];
      off_lds[n * 65 + 48 + m0 + r] = acc3[r];
    }
  }
  __syncthreads();

  // ---- bilinear sampling (matches reference's stack((off_h, off_w))) ----
  for (int i = 0; i < 4; ++i) {
    int j  = tid + 256 * i;   // 0..1023 : 64 px x 16 channel-groups
    int cg = j & 15;
    int m  = j >> 4;
    int w  = w0 + m;
    half4 outv;
#pragma unroll
    for (int q = 0; q < 4; ++q) {
      int c = cg * 4 + q;
      float off1 = off_lds[c * 65 + m];          // channels [0,64)   -> iy
      float off2 = off_lds[(c + 64) * 65 + m];   // channels [64,128) -> ix
      // ix = (off2 + gh(h) + 1)*63.5 = h + 63.5*off2 ; iy = w + 63.5*off1
      float ix = (float)h + 63.5f * off2;
      float iy = (float)w + 63.5f * off1;
      float x0f = floorf(ix), y0f = floorf(iy);
      float wx1 = ix - x0f, wy1 = iy - y0f;
      float wx0 = 1.0f - wx1, wy0 = 1.0f - wy1;
      int x0i = (int)x0f, y0i = (int)y0f;
      const float* img = x + (size_t)(b * 64 + c) * 16384;
      float v = gather_px(img, x0i,     y0i    ) * wy0 * wx0
              + gather_px(img, x0i + 1, y0i    ) * wy0 * wx1
              + gather_px(img, x0i,     y0i + 1) * wy1 * wx0
              + gather_px(img, x0i + 1, y0i + 1) * wy1 * wx1;
      outv[q] = (_Float16)v;
    }
    *(half4*)&xs[(((size_t)b * 128 + h) * 128 + w) * 64 + cg * 4] = outv;
  }
}

// ---------------------------------------------------------------------------
// Final conv: sampled (f16 NHWC) x wregt -> f32 NCHW out, + bias.
// WG = 4 waves; M-tile = 64, wave n owns out channels [16n,16n+16).
// ---------------------------------------------------------------------------
__global__ __launch_bounds__(128) void conv_out_kernel(
    const _Float16* __restrict__ xs,    // f16 NHWC sampled
    const _Float16* __restrict__ wregt, // [64][576]
    const float* __restrict__ bias,
    float* __restrict__ out)            // f32 NCHW
{
  __shared__ _Float16 atile[3 * 66 * 64] __attribute__((aligned(16)));

  const int tid  = threadIdx.x;
  const int wid  = tid >> 5;            // 0..3
  const int lane = tid & 31;
  const int l15  = lane & 15;
  const int lh   = lane >> 4;

  const int gid = blockIdx.x;
  const int wt  = gid & 1;
  const int h   = (gid >> 1) & 127;
  const int b   = gid >> 8;
  const int w0  = wt * 64;

  for (int t = tid; t < 3 * 66 * 16; t += 128) {
    int g = t & 15;
    int p = (t >> 4) % 66;
    int r = (t >> 4) / 66;
    int hh = h - 1 + r, ww = w0 - 1 + p;
    half4 v = {};
    if (hh >= 0 && hh < 128 && ww >= 0 && ww < 128)
      v = *(const half4*)&xs[(((size_t)b * 128 + hh) * 128 + ww) * 64 + g * 4];
    *(half4*)&atile[(r * 66 + p) * 64 + g * 4] = v;
  }
  __syncthreads();

  floatx8 acc0 = {}, acc1 = {}, acc2 = {}, acc3 = {};
  const int n0 = wid * 16;
  const _Float16* bptr = wregt + (size_t)(n0 + l15) * KTOT + 8 * lh;

  for (int s = 0; s < 18; ++s) {
    int khw = s >> 1;
    int kh = khw / 3, kw = khw - kh * 3;
    int cib = (s & 1) * 32 + 8 * lh;

    half8 blo = *(const half8*)(bptr + 32 * s);
    half8 bhi = *(const half8*)(bptr + 32 * s + 16);
    half16 Bf = __builtin_shufflevector(blo, bhi, 0, 1, 2, 3, 4, 5, 6, 7, 8,
                                        9, 10, 11, 12, 13, 14, 15);

    acc0 = __builtin_amdgcn_wmma_f32_16x16x32_f16(false, AFRAG(0), false, Bf,
                                                  (short)0, acc0, false, false);
    acc1 = __builtin_amdgcn_wmma_f32_16x16x32_f16(false, AFRAG(1), false, Bf,
                                                  (short)0, acc1, false, false);
    acc2 = __builtin_amdgcn_wmma_f32_16x16x32_f16(false, AFRAG(2), false, Bf,
                                                  (short)0, acc2, false, false);
    acc3 = __builtin_amdgcn_wmma_f32_16x16x32_f16(false, AFRAG(3), false, Bf,
                                                  (short)0, acc3, false, false);
  }

  const int n  = n0 + l15;
  const float bv = bias[n];
  const int m0 = 8 * lh;
  floatx8 v0 = acc0 + bv, v1 = acc1 + bv, v2 = acc2 + bv, v3 = acc3 + bv;
  size_t base = (((size_t)(b * 64 + n)) * 128 + h) * 128 + w0 + m0;
  *(floatx8*)&out[base +  0] = v0;
  *(floatx8*)&out[base + 16] = v1;
  *(floatx8*)&out[base + 32] = v2;
  *(floatx8*)&out[base + 48] = v3;
}

// ---------------------------------------------------------------------------
extern "C" void kernel_launch(void* const* d_in, const int* in_sizes, int n_in,
                              void* d_out, int out_size, void* d_ws,
                              size_t ws_size, hipStream_t stream)
{
  const float* x     = (const float*)d_in[0];
  const float* w_off = (const float*)d_in[1];
  const float* w_reg = (const float*)d_in[2];
  const float* b_reg = (const float*)d_in[3];
  float* out = (float*)d_out;

  char* ws = (char*)d_ws;
  const size_t XH_BYTES = (size_t)8 * 128 * 128 * 64 * 2;  // 16 MiB
  _Float16* xh    = (_Float16*)(ws);
  _Float16* xs    = (_Float16*)(ws + XH_BYTES);
  _Float16* wofft = (_Float16*)(ws + 2 * XH_BYTES);
  _Float16* wregt = (_Float16*)(ws + 2 * XH_BYTES + (size_t)128 * KTOT * 2);

  pack_weights<<<(128 * KTOT + 64 * KTOT + 255) / 256, 256, 0, stream>>>(
      w_off, w_reg, wofft, wregt);
  convert_x<<<(8 * 128 * 128 * 16) / 256, 256, 0, stream>>>(x, xh);
  conv_offset_sample<<<8 * 128 * 2, 256, 0, stream>>>(x, xh, wofft, xs);
  conv_out_kernel<<<8 * 128 * 2, 128, 0, stream>>>(xs, wregt, b_reg, out);
}
